// CustomSparseAttention_31018253812269
// MI455X (gfx1250) — matrix operand order repaired and checked by reading, not compiled, not run on hardware
//
#include <hip/hip_runtime.h>
#include <cstdint>

typedef __attribute__((ext_vector_type(16))) __bf16 v16bf;
typedef __attribute__((ext_vector_type(8)))  float  v8f;
typedef unsigned short u16;
typedef unsigned int   u32;
typedef unsigned long long u64;

#define B_DIM 2
#define T_DIM 2048
#define S_DIM 2048
#define E_DIM 512
#define H_NUM 16
#define HD    32
#define KDIM  512
#define NDIM  512
#define MDIM  4096   // B*T rows

struct MaskParams {
  u32 colmask[64];   // bit j of colmask[j>>5] : column j is a global token
  u64 chunkmask;     // bit c : 32-col chunk c contains a global column
};

__device__ __forceinline__ int imax(int a, int b){ return a > b ? a : b; }
__device__ __forceinline__ int imin(int a, int b){ return a < b ? a : b; }

__device__ __forceinline__ __bf16 f2bf(float f){ return (__bf16)f; }
__device__ __forceinline__ u16 bf_bits(float f){
  return __builtin_bit_cast(u16, (__bf16)f);
}

union BFV { uint4 u[2]; v16bf v; };

// 16 contiguous bf16 (unused; operands load via ld8_8)
__device__ __forceinline__ v16bf ld16(const u16* p){
  BFV x; const uint4* q = (const uint4*)p;
  x.u[0] = q[0]; x.u[1] = q[1];
  return x.v;
}
// bf16 halfs [0..7] and [16..23] relative to p (A-operand per-lane layout)
__device__ __forceinline__ v16bf ld8_8(const u16* p){
  BFV x;
  x.u[0] = *(const uint4*)p;
  x.u[1] = *(const uint4*)(p + 16);
  return x.v;
}
// fp32 -> bf16, floats [0..7] and [16..23] (A-operand)
__device__ __forceinline__ v16bf lda_f32(const float* p){
  float4 x0 = ((const float4*)p)[0], x1 = ((const float4*)p)[1];
  float4 x2 = ((const float4*)(p + 16))[0], x3 = ((const float4*)(p + 16))[1];
  v16bf v;
  v[0]=f2bf(x0.x); v[1]=f2bf(x0.y); v[2]=f2bf(x0.z); v[3]=f2bf(x0.w);
  v[4]=f2bf(x1.x); v[5]=f2bf(x1.y); v[6]=f2bf(x1.z); v[7]=f2bf(x1.w);
  v[8]=f2bf(x2.x); v[9]=f2bf(x2.y); v[10]=f2bf(x2.z); v[11]=f2bf(x2.w);
  v[12]=f2bf(x3.x); v[13]=f2bf(x3.y); v[14]=f2bf(x3.z); v[15]=f2bf(x3.w);
  return v;
}
__device__ __forceinline__ v8f wmma_bf16(v16bf a, v16bf b, v8f c){
  return __builtin_amdgcn_wmma_f32_16x16x32_bf16(false, a, false, b, (short)0, c,
                                                 false, false);
}

// ---------------------------------------------------------------------------
// Setup: fp32 -> bf16 bulk convert (weights), 8 elements per thread
// ---------------------------------------------------------------------------
__global__ __launch_bounds__(256) void cvt_bf16(const float* __restrict__ src,
                                                u16* __restrict__ dst, int n8)
{
  int idx = blockIdx.x * 256 + threadIdx.x;
  if (idx >= n8) return;
  const float4* p = (const float4*)src + (size_t)idx * 2;
  float4 a = p[0], b = p[1];
  union { uint4 u; u16 s[8]; } pk;
  pk.s[0]=bf_bits(a.x); pk.s[1]=bf_bits(a.y); pk.s[2]=bf_bits(a.z); pk.s[3]=bf_bits(a.w);
  pk.s[4]=bf_bits(b.x); pk.s[5]=bf_bits(b.y); pk.s[6]=bf_bits(b.z); pk.s[7]=bf_bits(b.w);
  ((uint4*)dst)[idx] = pk.u;
}

// ---------------------------------------------------------------------------
// Setup: transposed mask bitmap. maskT[col*64 + w] bit b = mask(i = 32w+b, col)
// Both the local window and the block-window union are contiguous row
// intervals per column, so each word is two range masks (branch-free).
// ---------------------------------------------------------------------------
__device__ __forceinline__ u32 range_bits(int lo, int hi /*inclusive*/, int base)
{
  const int l = imax(lo - base, 0);
  const int h = imin(hi - base, 31);
  if (l > h) return 0u;
  const u32 upper = (h >= 31) ? 0xFFFFFFFFu : ((1u << (h + 1)) - 1u);
  const u32 lower = (1u << l) - 1u;
  return upper & ~lower;
}

__global__ __launch_bounds__(256) void build_maskT(u32* __restrict__ maskT,
                                                   MaskParams mp)
{
  const int idx = blockIdx.x * 256 + threadIdx.x;   // 2048*64 = 131072 words
  const int col = idx >> 6, base = (idx & 63) << 5;
  const u32 g = (mp.colmask[col >> 5] >> (col & 31)) & 1u;
  // block windows covering this column: t in [clo, chi]; row union is the
  // contiguous interval [96*clo, 96*chi + 127] (windows overlap by 32 rows)
  const int clo = imax(0, (col - 64) / 96), chi = (col + 32) / 96;
  u32 bits = range_bits(col - 64, col + 64, base)          // sliding window
           | range_bits(96 * clo, 96 * chi + 127, base);   // block windows
  if (g) bits = 0xFFFFFFFFu;                               // global column
  maskT[idx] = bits;
}

// ---------------------------------------------------------------------------
// GEMM: out[m,n] = act( sum_k A[m,k] * W[n,k] + bias[n] ),  W pre-cvt bf16
// MODE 0: A fp32, out bf16 [M,N], scaled by scaling*temperature (Q)
// MODE 1: A fp32, out bf16 [M,N]                                  (K)
// MODE 2: A fp32, out bf16 transposed to [B,H,hd,S]               (V)
// MODE 3: A bf16, out fp32 [M,N]                                  (O)
// One wave computes a 32x32 tile; K loop in steps of 32 (bf16 WMMA).
// ---------------------------------------------------------------------------
template<int MODE>
__global__ __launch_bounds__(256) void gemm_wmma(
    const void* __restrict__ Aptr, const u16* __restrict__ Wb,
    const float* __restrict__ bias, void* __restrict__ Out,
    const float* __restrict__ tempPtr, float scaleConst)
{
  const int lane = threadIdx.x & 31;
  const int wid  = (blockIdx.x << 3) | (threadIdx.x >> 5);
  const int nt = wid & 15;          // NDIM/32 = 16 tiles
  const int mt = wid >> 4;          // MDIM/32 = 128 tiles
  const int m0 = mt << 5, n0 = nt << 5;
  const int ln = lane & 15, hi = lane >> 4;

  v8f c00 = {}, c01 = {}, c10 = {}, c11 = {};

  for (int kk = 0; kk < KDIM; kk += 32) {
    v16bf a0, a1;
    if (MODE == 3) {
      const u16* A = (const u16*)Aptr;
      const u16* p0 = A + (size_t)(m0 + ln) * KDIM + kk + hi * 8;
      a0 = ld8_8(p0);
      a1 = ld8_8(p0 + (size_t)16 * KDIM);
    } else {
      const float* A = (const float*)Aptr;
      a0 = lda_f32(A + (size_t)(m0 + ln) * KDIM + kk + hi * 8);
      a1 = lda_f32(A + (size_t)(m0 + 16 + ln) * KDIM + kk + hi * 8);
    }
    v16bf b0 = ld8_8(Wb + (size_t)(n0 + ln) * KDIM + kk + hi * 8);
    v16bf b1 = ld8_8(Wb + (size_t)(n0 + 16 + ln) * KDIM + kk + hi * 8);
    c00 = wmma_bf16(a0, b0, c00);
    c01 = wmma_bf16(a0, b1, c01);
    c10 = wmma_bf16(a1, b0, c10);
    c11 = wmma_bf16(a1, b1, c11);
  }

  float scl = 1.0f;
  if (MODE == 0) scl = scaleConst * tempPtr[0];

  v8f acc[2][2] = {{c00, c01}, {c10, c11}};
  #pragma unroll
  for (int mi = 0; mi < 2; mi++) {
    #pragma unroll
    for (int ni = 0; ni < 2; ni++) {
      const int n = n0 + ni * 16 + ln;
      const float bv = bias[n];
      if (MODE == 2) {
        // transposed store: rows (= s) are contiguous per lane -> packed b128
        const int mbase = m0 + mi * 16 + hi * 8;
        const int bb = mbase >> 11;           // / S_DIM
        const int sbase = mbase & (S_DIM - 1);
        const int hh = n >> 5, dd = n & (HD - 1);
        union { uint4 u; u16 s[8]; } pk;
        #pragma unroll
        for (int r = 0; r < 8; r++) pk.s[r] = bf_bits(acc[mi][ni][r] + bv);
        u16* vT = (u16*)Out;
        *(uint4*)(vT + (size_t)((bb * H_NUM + hh) * HD + dd) * S_DIM + sbase) = pk.u;
      } else {
        #pragma unroll
        for (int r = 0; r < 8; r++) {
          const int m = m0 + mi * 16 + hi * 8 + r;
          float v = acc[mi][ni][r] + bv;
          if (MODE == 0) v *= scl;
          if (MODE == 3) ((float*)Out)[(size_t)m * NDIM + n] = v;
          else           ((u16*)Out)[(size_t)m * NDIM + n] = bf_bits(v);
        }
      }
    }
  }
}

// ---------------------------------------------------------------------------
// Sparse flash attention: one wave per (b, h, 16-row query tile).
// S in 32-col chunks; fully-masked chunks skipped (uniform test).
// Reference value m = running max over RAW scores (exact softmax for any
// per-row-consistent m); mask applied by AND on packed bf16 P bits.
// Row sums accumulate through a P x ones WMMA.
// ---------------------------------------------------------------------------
__global__ __launch_bounds__(256) void attn_sparse(
    const u16* __restrict__ qb, const u16* __restrict__ kb,
    const u16* __restrict__ vT, const u32* __restrict__ maskT,
    u16* __restrict__ ctx, u64 chunkmask)
{
  __shared__ u16 pbuf[8][16 * 32];   // per-wave P tile (bf16, row-major 16x32)

  const int lane  = threadIdx.x & 31;
  const int wslot = threadIdx.x >> 5;
  const int wid   = (blockIdx.x << 3) | wslot;
  const int b   = wid >> 11;                 // 2048 waves per batch
  const int rem = wid & 2047;
  const int h   = rem >> 7;                  // 128 m-tiles per head
  const int i0  = (rem & 127) << 4;
  const int ln = lane & 15, hi = lane >> 4;

  // Q tile (A operand), loaded once and held for the whole S loop
  const u16* qp = qb + (size_t)(b * T_DIM + i0 + ln) * E_DIM + h * HD + hi * 8;
  const v16bf aq = ld8_8(qp);

  // ones B-operand for row-sum WMMA
  v16bf ones;
  #pragma unroll
  for (int i = 0; i < 16; i++) ones[i] = __builtin_bit_cast(__bf16, (u16)0x3F80);

  float mrun = 0.0f;                 // running reference (exact math, safe here)
  v8f o0 = {}, o1 = {}, ol = {};

  u16* myp = &pbuf[wslot][0];
  const int mword = i0 >> 5;                      // row word index in maskT
  const int mshift = (i0 & 31) + (hi << 3);       // bit of row (i0 + hi*8)

  const u16* kbase = kb + (size_t)(b * S_DIM + ln) * E_DIM + h * HD + hi * 8;
  const u16* vbase = vT + (size_t)((b * H_NUM + h) * HD + ln) * S_DIM + hi * 8;

  for (int sc = 0; sc < 64; sc++) {
    const int j0 = sc << 5;
    // ---- uniform chunk-skip test (mask is union of ranges) ----
    const bool loc = (j0 + 31 >= i0 - 64) && (j0 <= i0 + 79);
    const int tlo_r = imax(0, (i0 - 32) / 96), thi_r = (i0 + 15) / 96;
    const int tlo_c = imax(0, (j0 - 64) / 96), thi_c = (j0 + 63) / 96;
    const bool blk = imax(tlo_r, tlo_c) <= imin(thi_r, thi_c);
    const bool glb = (chunkmask >> sc) & 1ull;
    if (!(loc | blk | glb)) continue;

    // ---- scores: q (16x32) x k^T (32x16), two column tiles ----
    const u16* kp0 = kbase + (size_t)j0 * E_DIM;
    const v16bf bk0 = ld8_8(kp0);
    const v16bf bk1 = ld8_8(kp0 + (size_t)16 * E_DIM);
    // prefetch next chunk's K rows (each row == one 64B cacheline)
    __builtin_prefetch(kp0 + (size_t)32 * E_DIM, 0, 3);
    __builtin_prefetch(kp0 + (size_t)48 * E_DIM, 0, 3);

    v8f z = {};
    v8f s0 = wmma_bf16(aq, bk0, z);
    v8f s1 = wmma_bf16(aq, bk1, z);

    // ---- tile-wide max over RAW scores (no per-element selects) ----
    float mx = fmaxf(s0[0], s1[0]);
    #pragma unroll
    for (int r = 1; r < 8; r++) mx = fmaxf(mx, fmaxf(s0[r], s1[r]));
    mx = fmaxf(mx, __shfl_xor(mx, 1, 32));
    mx = fmaxf(mx, __shfl_xor(mx, 2, 32));
    mx = fmaxf(mx, __shfl_xor(mx, 4, 32));
    mx = fmaxf(mx, __shfl_xor(mx, 8, 32));
    mx = fmaxf(mx, __shfl_xor(mx, 16, 32));

    if (mx > mrun) {                 // wave-uniform: rescale rarely fires
      const float corr = __expf(mrun - mx);
      mrun = mx;
      #pragma unroll
      for (int r = 0; r < 8; r++) { o0[r] *= corr; o1[r] *= corr; ol[r] *= corr; }
    }

    // ---- mask from transposed bitmap: 1 word per column ----
    const int col0 = j0 + ln;
    const u32 mw0 = maskT[(size_t)col0 * 64 + mword] >> mshift;
    const u32 mw1 = maskT[(size_t)(col0 + 16) * 64 + mword] >> mshift;

    #pragma unroll
    for (int r = 0; r < 8; r++) {
      // masked entries -> AND with 0 on the bf16 bits (exact zero prob)
      const u16 k0 = (u16)(0u - ((mw0 >> r) & 1u));
      const u16 k1 = (u16)(0u - ((mw1 >> r) & 1u));
      const u16 pb0 = bf_bits(__expf(s0[r] - mrun)) & k0;
      const u16 pb1 = bf_bits(__expf(s1[r] - mrun)) & k1;
      const int rg = r + (hi << 3);
      myp[rg * 32 + ln]      = pb0;
      myp[rg * 32 + 16 + ln] = pb1;
    }

    // P (C layout) -> A layout via LDS (DS ops are in-order within a wave)
    const v16bf pa = ld8_8(myp + ln * 32 + hi * 8);

    // row sums: P x ones  (accumulates the softmax denominator per row)
    ol = wmma_bf16(pa, ones, ol);

    // V tiles from transposed layout: two 16B loads per lane ([0..7],[16..23])
    const u16* vp0 = vbase + j0;
    __builtin_prefetch(vp0 + 32, 0, 3);
    __builtin_prefetch(vp0 + (size_t)16 * S_DIM + 32, 0, 3);
    o0 = wmma_bf16(pa, ld8_8(vp0), o0);
    o1 = wmma_bf16(pa, ld8_8(vp0 + (size_t)16 * S_DIM), o1);
  }

  // normalize and store context bf16 [B,T,H,hd]
  #pragma unroll
  for (int r = 0; r < 8; r++) {
    const float inv = 1.0f / ol[r];
    const int t = i0 + r + (hi << 3);
    const size_t base = (size_t)(b * T_DIM + t) * E_DIM + h * HD + ln;
    ctx[base]      = bf_bits(o0[r] * inv);
    ctx[base + 16] = bf_bits(o1[r] * inv);
  }
}

// ---------------------------------------------------------------------------
// Host side
// ---------------------------------------------------------------------------
static MaskParams make_mask_params()
{
  MaskParams mp;
  for (int i = 0; i < 64; i++) mp.colmask[i] = 0u;
  mp.chunkmask = 0ull;
  // np.linspace(stride, S-stride, G).astype(int64), stride = 2048//17 = 120
  const double start = 120.0, stop = 2048.0 - 120.0;
  for (int i = 0; i < 16; i++) {
    double v = start + (stop - start) * (double)i / 15.0;
    long long g = (long long)v;                 // truncation, matches .long()
    mp.colmask[g >> 5] |= (1u << (g & 31));
    mp.chunkmask      |= (1ull << (g >> 5));    // chunk width == 32 == word
  }
  return mp;
}

extern "C" void kernel_launch(void* const* d_in, const int* in_sizes, int n_in,
                              void* d_out, int out_size, void* d_ws, size_t ws_size,
                              hipStream_t stream)
{
  (void)in_sizes; (void)n_in; (void)out_size; (void)ws_size;
  const float* query = (const float*)d_in[0];
  const float* key   = (const float*)d_in[1];
  const float* value = (const float*)d_in[2];
  const float* Wq = (const float*)d_in[3];
  const float* bq = (const float*)d_in[4];
  const float* Wk = (const float*)d_in[5];
  const float* bk = (const float*)d_in[6];
  const float* Wv = (const float*)d_in[7];
  const float* bv = (const float*)d_in[8];
  const float* Wo = (const float*)d_in[9];
  const float* bo = (const float*)d_in[10];
  const float* temp = (const float*)d_in[11];

  // workspace layout (u16 units):
  //   qb,kb [M,E] bf16; vT [B,H,hd,S] bf16; ctx [M,E] bf16;
  //   4 bf16 weights [N,K]; maskT 131072 u32
  u16* qb  = (u16*)d_ws;
  u16* kbf = qb  + (size_t)MDIM * E_DIM;
  u16* vTb = kbf + (size_t)MDIM * E_DIM;
  u16* ctx = vTb + (size_t)MDIM * E_DIM;
  u16* wqb = ctx + (size_t)MDIM * E_DIM;
  u16* wkb = wqb + (size_t)NDIM * KDIM;
  u16* wvb = wkb + (size_t)NDIM * KDIM;
  u16* wob = wvb + (size_t)NDIM * KDIM;
  u32* maskT = (u32*)(wob + (size_t)NDIM * KDIM);

  const float scaling = 0.1767766952966369f;   // hd^-0.5, hd = 32
  const MaskParams mp = make_mask_params();

  dim3 blk(256);
  // setup
  build_maskT<<<512, blk, 0, stream>>>(maskT, mp);
  const int w8 = (NDIM * KDIM) / 8;            // 32768 threads per weight
  cvt_bf16<<<128, blk, 0, stream>>>(Wq, wqb, w8);
  cvt_bf16<<<128, blk, 0, stream>>>(Wk, wkb, w8);
  cvt_bf16<<<128, blk, 0, stream>>>(Wv, wvb, w8);
  cvt_bf16<<<128, blk, 0, stream>>>(Wo, wob, w8);
  // projections
  gemm_wmma<0><<<256, blk, 0, stream>>>(query, wqb, bq, qb,  temp,   scaling);
  gemm_wmma<1><<<256, blk, 0, stream>>>(key,   wkb, bk, kbf, nullptr, 1.0f);
  gemm_wmma<2><<<256, blk, 0, stream>>>(value, wvb, bv, vTb, nullptr, 1.0f);
  // fused sparse flash attention
  attn_sparse<<<512, blk, 0, stream>>>(qb, kbf, vTb, maskT, ctx, mp.chunkmask);
  // output projection
  gemm_wmma<3><<<256, blk, 0, stream>>>(ctx, wob, bo, d_out, nullptr, 1.0f);
}
